// NTXentLoss_78408922955853
// MI455X (gfx1250) — matrix operand order repaired and checked by reading, not compiled, not run on hardware
//
#include <hip/hip_runtime.h>
#include <hip/hip_bf16.h>
#include <math.h>

// Problem constants (match reference)
#define N_ROWS 4096
#define DIM    512
#define TWO_N  8192
#define INV_T  10.0f   // 1 / temperature (temperature = 0.1)

// GEMM blocking
#define BLK        128            // block C tile: 128x128
#define BK         32             // k per stage (one WMMA k-step)
#define NKSTEP     (DIM / BK)     // 16 stages
#define LDS_STRIDE 40             // f16 per staged row (80 B: 64 B data + pad)

typedef __attribute__((ext_vector_type(16))) _Float16 v16h;
typedef __attribute__((ext_vector_type(8)))  _Float16 v8h;
typedef __attribute__((ext_vector_type(8)))  float    v8f;

// ---------------------------------------------------------------------------
// gfx1250 async global->LDS path (ASYNCcnt), with safe fallback.
// Builtin signature (from clang diagnostic): 
//   void __builtin_amdgcn_global_load_async_to_lds_b128(
//        int4 addrspace(1)* gsrc, int4 addrspace(3)* ldst,
//        const int offset, const int cpol)
// ---------------------------------------------------------------------------
#if __has_builtin(__builtin_amdgcn_global_load_async_to_lds_b128)
#define HAVE_ASYNC_LDS 1
typedef int v4i_vs __attribute__((vector_size(16)));
typedef __attribute__((address_space(1))) v4i_vs* gb128_t;
typedef __attribute__((address_space(3))) v4i_vs* lb128_t;
#if __has_builtin(__builtin_amdgcn_s_wait_asynccnt)
#define WAIT_ASYNC(n) __builtin_amdgcn_s_wait_asynccnt(n)
#else
#define WAIT_ASYNC(n) asm volatile("s_wait_asynccnt " #n ::: "memory")
#endif
#else
#define HAVE_ASYNC_LDS 0
#define WAIT_ASYNC(n)
#endif

// ---------------------------------------------------------------------------
// Reductions (wave32)
// ---------------------------------------------------------------------------
__device__ __forceinline__ float waveReduceSum32(float v) {
    #pragma unroll
    for (int m = 16; m >= 1; m >>= 1) v += __shfl_xor(v, m, 32);
    return v;
}

__device__ __forceinline__ float blockReduceSum(float v, float* sred) {
    const int lane = threadIdx.x & 31;
    const int w    = threadIdx.x >> 5;
    v = waveReduceSum32(v);
    if (lane == 0) sred[w] = v;
    __syncthreads();
    if (w == 0) {
        float x = (lane < 8) ? sred[lane] : 0.0f;
        x = waveReduceSum32(x);
        if (lane == 0) sred[0] = x;
    }
    __syncthreads();
    float total = sred[0];
    __syncthreads();   // allow buffer reuse by caller
    return total;
}

// ---------------------------------------------------------------------------
// Kernel 0: zero the neg[] accumulators and the positive-dot accumulator
// ---------------------------------------------------------------------------
__global__ void init_ws_kernel(float* __restrict__ neg, float* __restrict__ dotsum) {
    int i = blockIdx.x * 256 + threadIdx.x;
    if (i < TWO_N) neg[i] = 0.0f;
    if (i == 0)    *dotsum = 0.0f;
}

// ---------------------------------------------------------------------------
// Kernel 1: per-row L2 normalize -> f16 workspace; accumulate sum_i dot(z1_i,z2_i)
// ---------------------------------------------------------------------------
__global__ void prep_kernel(const float* __restrict__ z1,
                            const float* __restrict__ z2,
                            _Float16* __restrict__ znh,
                            float* __restrict__ dotsum) {
    __shared__ float red[8];
    const int i = blockIdx.x;       // row in [0, 2N)
    const int t = threadIdx.x;      // 0..255

    const float* src = (i < N_ROWS) ? (z1 + (size_t)i * DIM)
                                    : (z2 + (size_t)(i - N_ROWS) * DIM);
    const float x0 = src[t];
    const float x1 = src[t + 256];

    const float ss  = blockReduceSum(x0 * x0 + x1 * x1, red);
    const float inv = 1.0f / fmaxf(sqrtf(ss), 1e-12f);

    _Float16* dst = znh + (size_t)i * DIM;
    dst[t]       = (_Float16)(x0 * inv);
    dst[t + 256] = (_Float16)(x1 * inv);

    if (i < N_ROWS) {   // positive pair uses UN-normalized z1, z2 (faithful to ref)
        const float* y = z2 + (size_t)i * DIM;
        const float d  = blockReduceSum(x0 * y[t] + x1 * y[t + 256], red);
        if (t == 0) atomicAdd(dotsum, d);
    }
}

// ---------------------------------------------------------------------------
// Kernel 2: fused C = Zn*Zn^T -> exp(C/T) -> row sums.
// Block = 256 threads (8 waves), C tile 128x128. Double-buffered LDS staging of
// the A panel (128 rows x 32k) and B panel (128 cols x 32k), copied with
// async global->LDS ops when available. Each wave computes a 32x64 sub-tile:
// 2 M-tiles x 4 N-tiles = 8 WMMA accumulators, A/B fragments read from LDS.
// ---------------------------------------------------------------------------
__global__ void __launch_bounds__(256)
gemm_expsum_kernel(const _Float16* __restrict__ znh, float* __restrict__ neg) {
    __shared__ __align__(16) _Float16 sA[2][BLK * LDS_STRIDE];  // 2 x 10 KB
    __shared__ __align__(16) _Float16 sB[2][BLK * LDS_STRIDE];  // 2 x 10 KB

    const int tid  = threadIdx.x;
    const int lane = tid & 31;
    const int wave = tid >> 5;
    const int half = lane >> 4;      // 0: lanes 0-15, 1: lanes 16-31
    const int mrow = lane & 15;
    const int rowGroup = wave >> 1;  // 0..3 : 32 rows each
    const int colGroup = wave & 1;   // 0..1 : 64 cols each

    const int rowBase = blockIdx.y * BLK;
    const int colBase = blockIdx.x * BLK;

    // Cooperative staging: thread t owns one 64-byte row-slice per stage.
    // t < 128 -> A panel row t (Zn row rowBase+t); t >= 128 -> B panel row t-128.
    const int isB  = tid >> 7;
    const int srow = tid & 127;
    const _Float16* gsrc = znh + (size_t)((isB ? colBase : rowBase) + srow) * DIM;
    _Float16* dst0 = (isB ? sB[0] : sA[0]) + srow * LDS_STRIDE;
    _Float16* dst1 = (isB ? sB[1] : sA[1]) + srow * LDS_STRIDE;

    auto stage = [&](int buf, int kk) {
        const _Float16* g = gsrc + kk * BK;
        _Float16*       l = buf ? dst1 : dst0;
#if HAVE_ASYNC_LDS
        __builtin_amdgcn_global_load_async_to_lds_b128((gb128_t)g, (lb128_t)l,  0, 0);
        __builtin_amdgcn_global_load_async_to_lds_b128((gb128_t)g, (lb128_t)l, 16, 0);
        __builtin_amdgcn_global_load_async_to_lds_b128((gb128_t)g, (lb128_t)l, 32, 0);
        __builtin_amdgcn_global_load_async_to_lds_b128((gb128_t)g, (lb128_t)l, 48, 0);
#else
        #pragma unroll
        for (int c = 0; c < 4; ++c)
            *(v8h*)(l + c * 8) = *(const v8h*)(g + c * 8);
#endif
    };

    v8f acc[8] = {};   // acc[nt*2+mt]: 16x16 f32 tiles

    stage(0, 0);

    for (int kk = 0; kk < NKSTEP; ++kk) {
        const int cur = kk & 1;
        if (kk + 1 < NKSTEP) {
            stage(cur ^ 1, kk + 1);   // overlap next-stage copy with this compute
            WAIT_ASYNC(4);            // stage kk complete (in-order), kk+1 in flight
        } else {
            WAIT_ASYNC(0);
        }
        __syncthreads();              // publish stage kk to all waves

        const _Float16* lA = sA[cur];
        const _Float16* lB = sB[cur];

        // A fragments (16x32 ISA layout): lane<16: row, k=[0..7]++[16..23];
        // lane>=16: same rows, k shifted by 8.
        v16h afr[2];
        #pragma unroll
        for (int mt = 0; mt < 2; ++mt) {
            const int r = rowGroup * 32 + mt * 16 + mrow;
            const v8h a0 = *(const v8h*)(lA + r * LDS_STRIDE + half * 8);
            const v8h a1 = *(const v8h*)(lA + r * LDS_STRIDE + half * 8 + 16);
            #pragma unroll
            for (int e = 0; e < 8; ++e) { afr[mt][e] = a0[e]; afr[mt][e + 8] = a1[e]; }
        }

        // B fragments (32x16 ISA layout): lanes 0-15 hold col=lane, k=[0..7]++[16..23];
        // lanes 16-31 hold col=lane-16, k shifted by 8 (same order as A).
        v16h bfr[4];
        #pragma unroll
        for (int nt = 0; nt < 4; ++nt) {
            const int c = colGroup * 64 + nt * 16 + mrow;
            const v8h b0 = *(const v8h*)(lB + c * LDS_STRIDE + half * 8);
            const v8h b1 = *(const v8h*)(lB + c * LDS_STRIDE + half * 8 + 16);
            #pragma unroll
            for (int e = 0; e < 8; ++e) { bfr[nt][e] = b0[e]; bfr[nt][e + 8] = b1[e]; }
        }

        #pragma unroll
        for (int nt = 0; nt < 4; ++nt) {
            #pragma unroll
            for (int mt = 0; mt < 2; ++mt) {
                acc[nt * 2 + mt] = __builtin_amdgcn_wmma_f32_16x16x32_f16(
                    /*neg_a=*/false, afr[mt], /*neg_b=*/false, bfr[nt],
                    /*c_mod=*/(short)0, acc[nt * 2 + mt],
                    /*reuse_a=*/false, /*reuse_b=*/false);
            }
        }

        __syncthreads();              // retire buffer before it is re-staged
    }

    // ---- Epilogue: exp(c/T), row-sum, atomic accumulate.
    // C layout: VGPR r, lane n(0-15) = C[M=r][n]; lane n+16 = C[M=r+8][n]
    float rsum[2][8];
    #pragma unroll
    for (int mt = 0; mt < 2; ++mt)
        #pragma unroll
        for (int r = 0; r < 8; ++r) rsum[mt][r] = 0.0f;

    #pragma unroll
    for (int nt = 0; nt < 4; ++nt) {
        #pragma unroll
        for (int mt = 0; mt < 2; ++mt) {
            #pragma unroll
            for (int r = 0; r < 8; ++r) {
                rsum[mt][r] += __expf(acc[nt * 2 + mt][r] * INV_T);
            }
        }
    }

    #pragma unroll
    for (int mt = 0; mt < 2; ++mt) {
        #pragma unroll
        for (int r = 0; r < 8; ++r) {
            float v = rsum[mt][r];
            v += __shfl_xor(v, 1, 32);
            v += __shfl_xor(v, 2, 32);
            v += __shfl_xor(v, 4, 32);
            v += __shfl_xor(v, 8, 32);   // sum 16 lanes within each half
            if (mrow == 0) {
                atomicAdd(&neg[rowBase + rowGroup * 32 + mt * 16 + r + half * 8], v);
            }
        }
    }
}

// ---------------------------------------------------------------------------
// Kernel 3: loss = mean(log(S[i] - exp(1/T))) - mean(log pos)
// ---------------------------------------------------------------------------
__global__ void finalize_kernel(const float* __restrict__ neg,
                                const float* __restrict__ dotsum,
                                float* __restrict__ out) {
    __shared__ float red[8];
    const float selfTerm = __expf(INV_T);   // exp(1/T): diagonal contribution
    float s = 0.0f;
    for (int i = threadIdx.x; i < TWO_N; i += 256) {
        const float nv = neg[i] - selfTerm;
        s += logf(fmaxf(nv, 1e-30f));
    }
    s = blockReduceSum(s, red);
    if (threadIdx.x == 0) {
        const float logPosMean = (*dotsum) * INV_T / (float)N_ROWS;
        out[0] = s / (float)TWO_N - logPosMean;
    }
}

// ---------------------------------------------------------------------------
// Host-side launcher
// ---------------------------------------------------------------------------
extern "C" void kernel_launch(void* const* d_in, const int* in_sizes, int n_in,
                              void* d_out, int out_size, void* d_ws, size_t ws_size,
                              hipStream_t stream) {
    (void)in_sizes; (void)n_in; (void)out_size; (void)ws_size;

    const float* z1 = (const float*)d_in[0];
    const float* z2 = (const float*)d_in[1];
    // d_in[2] = temperature (compile-time constant 0.1 in the reference)
    float* out = (float*)d_out;

    _Float16* znh    = (_Float16*)d_ws;
    float*    neg    = (float*)((char*)d_ws + (size_t)TWO_N * DIM * sizeof(_Float16));
    float*    dotsum = neg + TWO_N;

    init_ws_kernel<<<(TWO_N + 255) / 256, 256, 0, stream>>>(neg, dotsum);
    prep_kernel<<<TWO_N, 256, 0, stream>>>(z1, z2, znh, dotsum);

    dim3 grid(TWO_N / BLK, TWO_N / BLK);   // 64 x 64 blocks
    gemm_expsum_kernel<<<grid, 256, 0, stream>>>(znh, neg);

    finalize_kernel<<<1, 256, 0, stream>>>(neg, dotsum, out);
}
